// RoPEAttention_41377714929748
// MI455X (gfx1250) — matrix operand order repaired and checked by reading, not compiled, not run on hardware
//
#include <hip/hip_runtime.h>
#include <hip/hip_bf16.h>

#define HIDDEN   2048
#define NHEADS   16
#define HD       128
#define SEQ      2048

typedef __attribute__((ext_vector_type(16))) __bf16 bf16x16;
typedef __attribute__((ext_vector_type(8)))  float  floatx8;

union FragBF16 {            // 16 bf16 = 8 VGPRs, loaded as two 16B chunks
    bf16x16 v;
    uint4   q[2];
};

__device__ __forceinline__ unsigned short f2bf(float f) {
    unsigned int u = __float_as_uint(f);
    unsigned int r = u + 0x7FFFu + ((u >> 16) & 1u);   // round-to-nearest-even
    return (unsigned short)(r >> 16);
}
__device__ __forceinline__ float bf2f(unsigned short h) {
    return __uint_as_float(((unsigned int)h) << 16);
}

__device__ __forceinline__ floatx8 wmma_bf16(const FragBF16& a, const FragBF16& b, floatx8 c) {
    return __builtin_amdgcn_wmma_f32_16x16x32_bf16(false, a.v, false, b.v,
                                                   (short)0, c, false, false);
}

// ---------------------------------------------------------------- convert f32 -> bf16
__global__ void cvt_f32_bf16(const float* __restrict__ in,
                             unsigned short* __restrict__ out, long long n4) {
    long long i = (long long)blockIdx.x * blockDim.x + threadIdx.x;
    if (i >= n4) return;
    const float4 v = *(const float4*)(in + i * 4);
    unsigned short o[4] = { f2bf(v.x), f2bf(v.y), f2bf(v.z), f2bf(v.w) };
    *(uint2*)(out + i * 4) = *(const uint2*)o;
}

// ---------------------------------------------------------------- bf16 GEMM  C = A * B^T + bias
// A: [M,K] bf16 row-major, Bm: [N,K] bf16 row-major (weight layout), bias f32[N]
// Ping-pong LDS double buffering: one barrier per 32-K stage, global loads for
// stage k+1 overlap the WMMAs of stage k.
// MODE 0: scatter bf16 into qkv buffer [3][B][h][S][HD]
// MODE 1: plain f32 output [M,N]
template<int MODE>
__global__ __launch_bounds__(256)
void gemm_bf16(const unsigned short* __restrict__ A,
               const unsigned short* __restrict__ Bm,
               const float* __restrict__ bias,
               void* __restrict__ Cout,
               int M, int N, int K, int Bsz, int S) {
    __shared__ unsigned short sA[2][128 * 40];   // 128x32 tiles, stride 40 (80B = 5*16B)
    __shared__ unsigned short sB[2][128 * 40];

    const int lane = threadIdx.x & 31;
    const int w    = threadIdx.x >> 5;          // 8 waves
    const int wr   = w >> 2;                    // 0..1  (64-row strip)
    const int wc   = w & 3;                     // 0..3  (32-col strip)
    const int fm   = lane & 15;
    const int kh   = lane >> 4;

    const int rowBase = blockIdx.y * 128;
    const int colBase = blockIdx.x * 128;

    // per-thread staging coordinates (two chunks of 8 bf16 per operand)
    const int r0 = threadIdx.x >> 2;
    const int c0 = (threadIdx.x & 3) * 8;
    const int r1 = (threadIdx.x + 256) >> 2;
    const int c1 = ((threadIdx.x + 256) & 3) * 8;

    uint4 ra0, ra1, rb0, rb1;
    auto loadStage = [&](int k0) {
        ra0 = *(const uint4*)(A  + (size_t)(rowBase + r0) * K + k0 + c0);
        ra1 = *(const uint4*)(A  + (size_t)(rowBase + r1) * K + k0 + c1);
        rb0 = *(const uint4*)(Bm + (size_t)(colBase + r0) * K + k0 + c0);
        rb1 = *(const uint4*)(Bm + (size_t)(colBase + r1) * K + k0 + c1);
    };
    auto storeStage = [&](int buf) {
        *(uint4*)(&sA[buf][r0 * 40 + c0]) = ra0;
        *(uint4*)(&sA[buf][r1 * 40 + c1]) = ra1;
        *(uint4*)(&sB[buf][r0 * 40 + c0]) = rb0;
        *(uint4*)(&sB[buf][r1 * 40 + c1]) = rb1;
    };

    floatx8 acc[4][2];
    #pragma unroll
    for (int i = 0; i < 4; ++i)
        #pragma unroll
        for (int j = 0; j < 2; ++j) acc[i][j] = (floatx8){};

    loadStage(0);
    storeStage(0);
    __syncthreads();

    int cur = 0;
    for (int k0 = 0; k0 < K; k0 += 32) {
        const bool hasNext = (k0 + 32 < K);
        if (hasNext) loadStage(k0 + 32);          // overlaps WMMAs below

        FragBF16 af[4];
        #pragma unroll
        for (int i = 0; i < 4; ++i) {
            int base = (wr * 64 + i * 16 + fm) * 40 + kh * 8;   // K chunks 8*kh, 16+8*kh
            af[i].q[0] = *(const uint4*)(&sA[cur][base]);
            af[i].q[1] = *(const uint4*)(&sA[cur][base + 16]);
        }
        FragBF16 bf[2];
        #pragma unroll
        for (int j = 0; j < 2; ++j) {
            int base = (wc * 32 + j * 16 + fm) * 40 + kh * 8;   // K chunks 8*kh, 16+8*kh
            bf[j].q[0] = *(const uint4*)(&sB[cur][base]);
            bf[j].q[1] = *(const uint4*)(&sB[cur][base + 16]);
        }
        #pragma unroll
        for (int i = 0; i < 4; ++i)
            #pragma unroll
            for (int j = 0; j < 2; ++j)
                acc[i][j] = wmma_bf16(af[i], bf[j], acc[i][j]);

        if (hasNext) storeStage(cur ^ 1);
        __syncthreads();
        cur ^= 1;
    }

    // ---- epilogue ----
    const int n  = lane & 15;
    const int mh = lane >> 4;
    #pragma unroll
    for (int i = 0; i < 4; ++i) {
        #pragma unroll
        for (int jj = 0; jj < 2; ++jj) {
            int col  = colBase + wc * 32 + jj * 16 + n;
            int row0 = rowBase + wr * 64 + i * 16 + mh * 8;
            float bv = bias[col];
            #pragma unroll
            for (int j = 0; j < 8; ++j) {
                float val = acc[i][jj][j] + bv;
                int row = row0 + j;
                if constexpr (MODE == 0) {
                    int g = col >> 11, h = (col >> 7) & 15, d = col & 127;
                    int b = row / S,  s = row - b * S;
                    long long off = ((((long long)g * Bsz + b) * NHEADS + h) * S + s) * HD + d;
                    ((unsigned short*)Cout)[off] = f2bf(val);
                } else {
                    ((float*)Cout)[(size_t)row * N + col] = val;
                }
            }
        }
    }
}

// ---------------------------------------------------------------- RoPE on q,k planes (in-place, bf16)
__global__ void rope_kernel(unsigned short* __restrict__ qkv, int Bsz, int S, long long total) {
    long long idx = (long long)blockIdx.x * blockDim.x + threadIdx.x;
    if (idx >= total) return;                       // total = B*S*16*64
    int d = idx & 63;
    int h = (int)((idx >> 6) & 15);
    long long token = idx >> 10;
    int b = (int)(token / S), s = (int)(token - (long long)b * S);
    float inv_freq = __expf((float)d * -0.14391156f);   // 10000^(-d/64)
    float ang = (float)s * inv_freq;
    float c, sn;
    __sincosf(ang, &sn, &c);
    #pragma unroll
    for (int g = 0; g < 2; ++g) {                   // 0=q, 1=k
        long long base = ((((long long)g * Bsz + b) * NHEADS + h) * S + s) * HD;
        float x1 = bf2f(qkv[base + d]);
        float x2 = bf2f(qkv[base + 64 + d]);
        qkv[base + d]      = f2bf(x1 * c - x2 * sn);
        qkv[base + 64 + d] = f2bf(x2 * c + x1 * sn);
    }
}

// ---------------------------------------------------------------- flash attention (bf16 WMMA, f32 softmax)
// one block = (b, h, 64 q-rows); 4 waves, 16 q-rows per wave; 64 keys per chunk.
// Row-sums are accumulated as a 9th WMMA tile (P x Ones), so the only cross-lane
// traffic is the 4-stage row-max reduction once per 64 keys.
__global__ __launch_bounds__(128)
void attn_kernel(const unsigned short* __restrict__ qkv,
                 unsigned short* __restrict__ aout, int Bsz, int S) {
    __shared__ unsigned short sK[64 * 136];   // [key][hd]  stride 136 (272B = 17*16B)
    __shared__ unsigned short sV[128 * 72];   // [hd][key]  transposed, stride 72 (144B = 9*16B)
    __shared__ unsigned short sP[64 * 72];    // per-wave 16x64 P tiles, stride 72

    const int lane = threadIdx.x & 31;
    const int w    = threadIdx.x >> 5;
    const int fm   = lane & 15;
    const int kh   = lane >> 4;
    const int n    = fm;            // C-layout column
    const int mh   = kh;            // C-layout row half

    const int b = blockIdx.z, h = blockIdx.y;
    const size_t plane = (size_t)S * HD;
    const unsigned short* qp = qkv + (((size_t)0 * Bsz + b) * NHEADS + h) * plane;
    const unsigned short* kp = qkv + (((size_t)1 * Bsz + b) * NHEADS + h) * plane;
    const unsigned short* vp = qkv + (((size_t)2 * Bsz + b) * NHEADS + h) * plane;

    const int qrow0 = blockIdx.x * 64 + w * 16;

    // Q fragments resident in registers: 16 rows x 128 hd = 4 A-frags
    FragBF16 qf[4];
    #pragma unroll
    for (int k = 0; k < 4; ++k) {
        size_t base = (size_t)(qrow0 + fm) * HD + k * 32 + kh * 8;
        qf[k].q[0] = *(const uint4*)(qp + base);
        qf[k].q[1] = *(const uint4*)(qp + base + 16);
    }

    FragBF16 ones;                                 // bf16 1.0 replicated
    ones.q[0] = uint4{0x3f803f80u, 0x3f803f80u, 0x3f803f80u, 0x3f803f80u};
    ones.q[1] = ones.q[0];

    floatx8 o[8];
    #pragma unroll
    for (int t = 0; t < 8; ++t) o[t] = (floatx8){};
    floatx8 lacc = (floatx8){};                    // row-sum accumulator tile
    float mi[8];
    #pragma unroll
    for (int j = 0; j < 8; ++j) mi[j] = -1e30f;
    const float scale = 0.08838834764831845f;      // 1/sqrt(128)

    const int srow  = threadIdx.x >> 1;            // staging: 0..63
    const int shalf = (threadIdx.x & 1) * 64;      // staging: 0 or 64

    for (int kc = 0; kc < S; kc += 64) {
        // ---- stage K (natural) and V (transposed) ----
        #pragma unroll
        for (int i = 0; i < 8; ++i) {
            uint4 kv = *(const uint4*)(kp + (size_t)(kc + srow) * HD + shalf + i * 8);
            *(uint4*)(sK + srow * 136 + shalf + i * 8) = kv;
        }
        #pragma unroll
        for (int i = 0; i < 8; ++i) {
            uint4 vv = *(const uint4*)(vp + (size_t)(kc + srow) * HD + shalf + i * 8);
            const unsigned short* e = (const unsigned short*)&vv;
            #pragma unroll
            for (int c = 0; c < 8; ++c)
                sV[(shalf + i * 8 + c) * 72 + srow] = e[c];
        }
        if (kc + 64 < S) {                          // gfx1250 global_prefetch_b8
            __builtin_prefetch(kp + (size_t)(kc + 64 + srow) * HD + shalf, 0, 3);
            __builtin_prefetch(vp + (size_t)(kc + 64 + srow) * HD + shalf, 0, 3);
        }
        __syncthreads();

        // ---- scores: S = Q * K^T  (four 16x16 tiles over 64 keys) ----
        floatx8 s[4];
        #pragma unroll
        for (int kt = 0; kt < 4; ++kt) s[kt] = (floatx8){};
        #pragma unroll
        for (int k = 0; k < 4; ++k) {
            #pragma unroll
            for (int kt = 0; kt < 4; ++kt) {
                FragBF16 kf;
                int kb = (kt * 16 + fm) * 136 + k * 32 + kh * 8;
                kf.q[0] = *(const uint4*)(sK + kb);
                kf.q[1] = *(const uint4*)(sK + kb + 16);
                s[kt] = wmma_bf16(qf[k], kf, s[kt]);
            }
        }

        // ---- online softmax (rows j+8*mh; 16-lane max reduction within half) ----
        #pragma unroll
        for (int j = 0; j < 8; ++j) {
            float a0 = s[0][j] * scale, a1 = s[1][j] * scale;
            float a2 = s[2][j] * scale, a3 = s[3][j] * scale;
            float tm = fmaxf(fmaxf(a0, a1), fmaxf(a2, a3));
            #pragma unroll
            for (int off = 8; off > 0; off >>= 1)
                tm = fmaxf(tm, __shfl_xor(tm, off, 32));
            float mn = fmaxf(mi[j], tm);
            float al = __expf(mi[j] - mn);
            mi[j] = mn;
            float p0 = __expf(a0 - mn), p1 = __expf(a1 - mn);
            float p2 = __expf(a2 - mn), p3 = __expf(a3 - mn);
            lacc[j] *= al;
            #pragma unroll
            for (int t = 0; t < 8; ++t) o[t][j] *= al;
            int pb = (w * 16 + mh * 8 + j) * 72 + n;    // C -> A re-layout via LDS
            sP[pb]      = f2bf(p0);
            sP[pb + 16] = f2bf(p1);
            sP[pb + 32] = f2bf(p2);
            sP[pb + 48] = f2bf(p3);
        }

        // ---- O += P * V ; lacc += P * Ones  (intra-wave DS ordering covers sP RAW)
        #pragma unroll
        for (int ks = 0; ks < 2; ++ks) {
            FragBF16 pf;
            int pb = (w * 16 + fm) * 72 + ks * 32 + kh * 8;
            pf.q[0] = *(const uint4*)(sP + pb);
            pf.q[1] = *(const uint4*)(sP + pb + 16);
            #pragma unroll
            for (int t = 0; t < 8; ++t) {
                FragBF16 vf;
                int vb = (t * 16 + fm) * 72 + ks * 32 + kh * 8;
                vf.q[0] = *(const uint4*)(sV + vb);
                vf.q[1] = *(const uint4*)(sV + vb + 16);
                o[t] = wmma_bf16(pf, vf, o[t]);
            }
            lacc = wmma_bf16(pf, ones, lacc);
        }
        __syncthreads();
    }

    // ---- normalize and store [B,S,h,HD] bf16 (l is lane-local via lacc) ----
    float inv[8];
    #pragma unroll
    for (int j = 0; j < 8; ++j) inv[j] = 1.0f / lacc[j];
    #pragma unroll
    for (int t = 0; t < 8; ++t) {
        #pragma unroll
        for (int j = 0; j < 8; ++j) {
            int qrow = qrow0 + mh * 8 + j;
            long long token = (long long)b * S + qrow;
            int col = h * HD + t * 16 + n;
            aout[token * HIDDEN + col] = f2bf(o[t][j] * inv[j]);
        }
    }
}

// ----------------------------------------------------------------
extern "C" void kernel_launch(void* const* d_in, const int* in_sizes, int n_in,
                              void* d_out, int out_size, void* d_ws, size_t ws_size,
                              hipStream_t stream) {
    const float* x     = (const float*)d_in[0];
    const float* w_qkv = (const float*)d_in[1];
    const float* b_qkv = (const float*)d_in[2];
    const float* w_out = (const float*)d_in[3];
    const float* b_out = (const float*)d_in[4];

    const int S   = SEQ;
    const int tok = in_sizes[0] / HIDDEN;      // B*S = 8192
    const int Bsz = tok / S;                   // 4

    char* ws = (char*)d_ws;
    unsigned short* xb    = (unsigned short*)(ws);                               // 32 MB
    unsigned short* wqkvb = (unsigned short*)(ws + (size_t)tok * HIDDEN * 2);    // 24 MB
    unsigned short* woutb = (unsigned short*)((char*)wqkvb + (size_t)3 * HIDDEN * HIDDEN * 2);
    unsigned short* qkvb  = (unsigned short*)((char*)woutb + (size_t)HIDDEN * HIDDEN * 2);
    unsigned short* attnb = (unsigned short*)((char*)qkvb + (size_t)tok * 3 * HIDDEN * 2);

    // 1) fp32 -> bf16 conversions
    {
        long long n4 = (long long)tok * HIDDEN / 4;
        cvt_f32_bf16<<<(unsigned)((n4 + 255) / 256), 256, 0, stream>>>(x, xb, n4);
        n4 = (long long)3 * HIDDEN * HIDDEN / 4;
        cvt_f32_bf16<<<(unsigned)((n4 + 255) / 256), 256, 0, stream>>>(w_qkv, wqkvb, n4);
        n4 = (long long)HIDDEN * HIDDEN / 4;
        cvt_f32_bf16<<<(unsigned)((n4 + 255) / 256), 256, 0, stream>>>(w_out, woutb, n4);
    }

    // 2) QKV projection + scatter to [3][B][h][S][HD]
    gemm_bf16<0><<<dim3(3 * HIDDEN / 128, tok / 128), 256, 0, stream>>>(
        xb, wqkvb, b_qkv, qkvb, tok, 3 * HIDDEN, HIDDEN, Bsz, S);

    // 3) RoPE on q,k
    {
        long long total = (long long)tok * NHEADS * 64;
        rope_kernel<<<(unsigned)((total + 255) / 256), 256, 0, stream>>>(qkvb, Bsz, S, total);
    }

    // 4) flash attention
    attn_kernel<<<dim3(S / 64, NHEADS, Bsz), 128, 0, stream>>>(qkvb, attnb, Bsz, S);

    // 5) output projection -> f32 d_out
    gemm_bf16<1><<<dim3(HIDDEN / 128, tok / 128), 256, 0, stream>>>(
        attnb, woutb, b_out, d_out, tok, HIDDEN, HIDDEN, Bsz, S);
}
